// MST11_12463995093524
// MI455X (gfx1250) — matrix operand order repaired and checked by reading, not compiled, not run on hardware
//
#include <hip/hip_runtime.h>
#include <hip/hip_bf16.h>
#include <math.h>
#include <stdint.h>

// =====================================================================
// MST (spectral transformer U-Net) on gfx1250 / MI455X.
//  - Activations: bf16 NHWC, channel dim padded to multiple of 32.
//  - Weights pre-packed once per launch to bf16 [tap][Np][Kp], zero pad.
//  - Implicit-GEMM conv kernel (templated on mode for straight-line
//    inner loops): double-buffered async global->LDS (b128) staging
//    pipelined against v_wmma_f32_16x16x32_bf16, fp32 accumulate,
//    fused bias/GELU/leaky/residual epilogue.
//  - Irregular ops (depthwise 3x3, LN, L2-norm, 31x31 channel attention)
//    stay VALU: bandwidth-bound.
// Workspace: ~400 MB carved with a bump allocator.
// =====================================================================

typedef unsigned short u16;   // raw bf16 bits
typedef __bf16 bf16_t;
typedef bf16_t v16bf __attribute__((ext_vector_type(16)));
typedef float  v8f   __attribute__((ext_vector_type(8)));

union FragBF { v16bf v; unsigned int u[8]; };
union FragF  { v8f   v; float        f[8]; };

__device__ inline u16 f2bf(float x) {
  union { float f; unsigned int u; } c; c.f = x;
  unsigned int r = c.u + 0x7FFFu + ((c.u >> 16) & 1u);   // RNE
  return (u16)(r >> 16);
}
__device__ inline float bf2f(u16 h) {
  union { float f; unsigned int u; } c; c.u = ((unsigned int)h) << 16;
  return c.f;
}
__device__ inline float gelu_exact(float v) {
  return 0.5f * v * (1.0f + erff(v * 0.70710678118654752f));
}

// async 16B global -> LDS (ISA 15.18.3, tracked by ASYNCcnt)
__device__ inline void async_cp16(unsigned lds_off, const u16* g) {
  asm volatile("global_load_async_to_lds_b128 %0, %1, off"
               :: "v"(lds_off), "v"(g) : "memory");
}
__device__ inline void wait_async_all() {
  asm volatile("s_wait_asynccnt 0x0" ::: "memory");
}

#define TM 64
#define TN 32
#define TK 32
#define ASTR 40                   // LDS row stride in u16 (16B aligned rows)
#define BUFSZ ((TM + TN) * ASTR)  // one double-buffer slot (u16 units)

// ---------------------------------------------------------------------
// MODE 0 (FAST):  1x1 / stride1 / pad0 matmul, rows contiguous.
// MODE 1 (CONV):  kh x kw conv, stride/pad, accumulate over taps.
// MODE 2 (TRANS): convT with kernel==stride; per-tap GEMM + scatter.
// Requires: Bn*Mh*Mw % 64 == 0, Cin_pad % 32 == 0, Kp/Np padded mult 32.
// ---------------------------------------------------------------------
template <int MODE>
__global__ void __launch_bounds__(128)
conv_gemm_wmma(const u16* __restrict__ in, const u16* __restrict__ wp,
               const float* __restrict__ bias, const u16* __restrict__ res,
               u16* __restrict__ out,
               int Hi, int Wi, int Cin, int Cin_pad,
               int Ho, int Wo, int Cout, int Cstride, int coff,
               int kh, int kw, int stride, int pad,
               int Kp, int Np, int act)
{
  __shared__ u16 smem[2 * BUFSZ];

  const int Mh = (MODE == 2) ? Hi : Ho;
  const int Mw = (MODE == 2) ? Wi : Wo;
  const int MhMw = Mh * Mw;

  const int m0 = blockIdx.x * TM;
  const int n0 = blockIdx.y * TN;

  const int tid  = threadIdx.x;
  const int lane = tid & 31;
  const int wave = tid >> 5;
  const int wm   = wave & 1;        // M half: rows wm*32..+32
  const int wn   = wave >> 1;       // N half: cols wn*16..+16
  const int lrow = lane & 15;
  const int lgrp = lane >> 4;

  const int ntaps = kh * kw;
  const int nk = (Cin + TK - 1) / TK;
  const int nt = (MODE == 0) ? nk : ntaps * nk;

  // ---- per-thread invariants for the tile loaders ----
  int amm[2], akk[2], apixb[2], aiyb[2], aixb[2];
  const u16* abase[2];
  #pragma unroll
  for (int cc = 0; cc < 2; ++cc) {
    const int c = tid + cc * 128;
    amm[cc] = c >> 2;
    akk[cc] = (c & 3) << 3;
    const int row = m0 + amm[cc];          // always valid: M % 64 == 0
    const int b = row / MhMw;
    const int rem = row - b * MhMw;
    const int oy = rem / Mw, ox = rem - oy * Mw;
    if (MODE == 0) {
      abase[cc] = in + (long)row * Cin_pad + akk[cc];
    } else if (MODE == 2) {
      abase[cc] = in + (((long)b * Hi + oy) * Wi + ox) * Cin_pad + akk[cc];
    } else {
      apixb[cc] = b * Hi;                   // row block in image space
      aiyb[cc]  = oy * stride - pad;
      aixb[cc]  = ox * stride - pad;
      abase[cc] = in + akk[cc];
    }
  }
  // B: one chunk per thread
  const int bnn  = tid >> 2;
  const int bkk  = (tid & 3) << 3;
  const u16* bbase = wp + (long)(n0 + bnn) * Kp + bkk;
  const long btap  = (long)Np * Kp;

  // issue all async loads / zero-stores for flat tile f into buffer `sel`
  auto issue = [&](int f, int sel) {
    int tap, k0;
    if (MODE == 0) { tap = 0; k0 = f * TK; }
    else { tap = f / nk; k0 = (f - tap * nk) * TK; }
    u16* Ab = smem + sel * BUFSZ;
    u16* Bb = Ab + TM * ASTR;
    #pragma unroll
    for (int cc = 0; cc < 2; ++cc) {
      const unsigned lofs =
          (unsigned)(uintptr_t)(Ab + amm[cc] * ASTR + akk[cc]);
      if (MODE == 1) {
        const int dy = tap / kw, dx = tap % kw;
        const int iy = aiyb[cc] + dy, ix = aixb[cc] + dx;
        if (iy >= 0 && iy < Hi && ix >= 0 && ix < Wi) {
          async_cp16(lofs, abase[cc] +
                     (((long)apixb[cc] + iy) * Wi + ix) * Cin_pad + k0);
        } else {
          uint4 z = {0u, 0u, 0u, 0u};
          *(uint4*)(Ab + amm[cc] * ASTR + akk[cc]) = z;
        }
      } else {
        async_cp16(lofs, abase[cc] + k0);
      }
    }
    const unsigned lofsB = (unsigned)(uintptr_t)(Bb + bnn * ASTR + bkk);
    async_cp16(lofsB, bbase + btap * tap + k0);
  };

  // ---- epilogue helper ----
  auto epilogue = [&](const FragF* acc, int tap) {
    const int dyT = (MODE == 2) ? tap / kw : 0;
    const int dxT = (MODE == 2) ? tap % kw : 0;
    #pragma unroll
    for (int sm = 0; sm < 2; ++sm) {
      #pragma unroll
      for (int j = 0; j < 8; ++j) {
        const int ml = wm * 32 + sm * 16 + (j + lgrp * 8);  // C/D: M = j + 8g
        const int nl = wn * 16 + lrow;                      // C/D: N = lane%16
        const int row = m0 + ml, o = n0 + nl;
        if (o < Cout) {
          float v = acc[sm].f[j];
          if (bias) v += bias[o];
          if (act == 1)      v = gelu_exact(v);
          else if (act == 2) v = (v > 0.f) ? v : 0.01f * v;
          const int b = row / MhMw;
          const int rem = row - b * MhMw;
          const int oy = rem / Mw, ox = rem - oy * Mw;
          long oi;
          if (MODE == 2) {
            const int yy = oy * stride + dyT, xx = ox * stride + dxT;
            oi = (((long)b * Ho + yy) * Wo + xx) * Cstride + coff + o;
          } else {
            oi = (((long)b * Ho + oy) * Wo + ox) * Cstride + coff + o;
          }
          if (res) v += bf2f(res[oi]);
          out[oi] = f2bf(v);
        }
      }
    }
  };

  // ---- software pipeline: async stage tile f+1 while computing tile f ----
  FragF acc[2];
  #pragma unroll
  for (int e = 0; e < 8; ++e) { acc[0].f[e] = 0.f; acc[1].f[e] = 0.f; }

  issue(0, 0);
  for (int f = 0; f < nt; ++f) {
    const int sel = f & 1;
    wait_async_all();       // tile f resident in LDS
    __syncthreads();        // visible to all waves; prior reads of buf done
    if (f + 1 < nt) issue(f + 1, sel ^ 1);   // overlaps with compute below

    const u16* Ab = smem + sel * BUFSZ;
    const u16* Bb = Ab + TM * ASTR;

    FragBF bf;
    {
      const u16* bp = Bb + (wn * 16 + lrow) * ASTR;
      #pragma unroll
      for (int j = 0; j < 8; ++j)                    // B: K = g*16 + 2j + h
        bf.u[j] = *(const unsigned int*)(bp + lgrp * 16 + j * 2);
    }
    #pragma unroll
    for (int sm = 0; sm < 2; ++sm) {
      FragBF af;
      const u16* ap = Ab + (wm * 32 + sm * 16 + lrow) * ASTR;
      #pragma unroll
      for (int j = 0; j < 8; ++j) {                  // A: K = g*16 + 2j + h
        const int kb = lgrp * 16 + j * 2;
        af.u[j] = *(const unsigned int*)(ap + kb);
      }
      acc[sm].v = __builtin_amdgcn_wmma_f32_16x16x32_bf16(
          false, af.v, false, bf.v, (short)0, acc[sm].v, false, false);
    }

    if (MODE == 2 && ((f + 1) % nk == 0)) {          // tap finished: scatter
      epilogue(acc, f / nk);
      #pragma unroll
      for (int e = 0; e < 8; ++e) { acc[0].f[e] = 0.f; acc[1].f[e] = 0.f; }
    }
  }
  if (MODE != 2) epilogue(acc, 0);
}

// ---------------------------------------------------------------------
// Weight pre-pack: fp32 (i*w_si + o*w_so + tap) -> bf16 [tap][Np][Kp]
// ---------------------------------------------------------------------
__global__ void pack_weight_k(const float* __restrict__ w, u16* __restrict__ wp,
                              int K, int N, int Kp, int Np, int taps,
                              int w_si, int w_so)
{
  long idx = (long)blockIdx.x * blockDim.x + threadIdx.x;
  long total = (long)taps * Np * Kp;
  if (idx >= total) return;
  int k = (int)(idx % Kp);
  long r = idx / Kp;
  int n = (int)(r % Np);
  int tap = (int)(r / Np);
  float v = (k < K && n < N) ? w[(long)k * w_si + (long)n * w_so + tap] : 0.f;
  wp[idx] = f2bf(v);
}

// ---------------------------------------------------------------------
// Depthwise 3x3 (pad 1), fused GELU + up to two adds. bf16 activations.
// ---------------------------------------------------------------------
__global__ void dwconv3x3_k(const u16* __restrict__ in, const float* __restrict__ w,
                            const u16* __restrict__ add1, const u16* __restrict__ add2,
                            u16* __restrict__ out,
                            int H, int W, int C, int Cp, int act)
{
  long idx = (long)blockIdx.x * blockDim.x + threadIdx.x;
  long total = (long)4 * H * W * Cp;
  if (idx >= total) return;
  int c = (int)(idx % Cp);
  long r = idx / Cp;
  if (c >= C) return;
  int x = (int)(r % W); long r2 = r / W;
  int y = (int)(r2 % H); int b = (int)(r2 / H);
  float s = 0.f;
  #pragma unroll
  for (int dy = 0; dy < 3; ++dy)
    #pragma unroll
    for (int dx = 0; dx < 3; ++dx) {
      int iy = y + dy - 1, ix = x + dx - 1;
      if (iy >= 0 && iy < H && ix >= 0 && ix < W)
        s += bf2f(in[(((long)b * H + iy) * W + ix) * Cp + c]) * w[c * 9 + dy * 3 + dx];
    }
  if (act == 1) s = gelu_exact(s);
  long oi = r * Cp + c;
  if (add1) s += bf2f(add1[oi]);
  if (add2) s += bf2f(add2[oi]);
  out[oi] = f2bf(s);
}

// LayerNorm over channels, eps 1e-5
__global__ void layernorm_k(const u16* __restrict__ in, const float* __restrict__ g,
                            const float* __restrict__ bta, u16* __restrict__ out,
                            long rows, int C, int Cp)
{
  long r = (long)blockIdx.x * blockDim.x + threadIdx.x;
  if (r >= rows) return;
  const u16* p = in + r * Cp;
  float mu = 0.f;
  for (int c = 0; c < C; ++c) mu += bf2f(p[c]);
  mu /= (float)C;
  float var = 0.f;
  for (int c = 0; c < C; ++c) { float d = bf2f(p[c]) - mu; var += d * d; }
  var /= (float)C;
  float rs = rsqrtf(var + 1e-5f);
  u16* q = out + r * Cp;
  for (int c = 0; c < C; ++c) q[c] = f2bf((bf2f(p[c]) - mu) * rs * g[c] + bta[c]);
}

// inverse L2 norm over n tokens per (batch, channel)
__global__ void l2norm_k(const u16* __restrict__ in, float* __restrict__ invn,
                         long n, int C, int Cp)
{
  int b = blockIdx.x / C, c = blockIdx.x % C;
  __shared__ float red[256];
  float s = 0.f;
  for (long r = threadIdx.x; r < n; r += 256) {
    float v = bf2f(in[((long)b * n + r) * Cp + c]);
    s += v * v;
  }
  red[threadIdx.x] = s; __syncthreads();
  for (int st = 128; st > 0; st >>= 1) {
    if (threadIdx.x < st) red[threadIdx.x] += red[threadIdx.x + st];
    __syncthreads();
  }
  if (threadIdx.x == 0)
    invn[b * C + c] = 1.f / fmaxf(sqrtf(red[0]), 1e-12f);
}

__global__ void scale_k(u16* __restrict__ q, const float* __restrict__ invn,
                        long n, int C, int Cp)
{
  long idx = (long)blockIdx.x * blockDim.x + threadIdx.x;
  long total = (long)4 * n * Cp;
  if (idx >= total) return;
  int c = (int)(idx % Cp);
  if (c >= C) return;
  long r = idx / Cp;
  int b = (int)(r / n);
  q[idx] = f2bf(bf2f(q[idx]) * invn[b * C + c]);
}

// attn[b,h,d,e] = rescale[h] * sum_n khat[.,d] * qhat[.,e]   (fp32 out)
__global__ void attn_k(const u16* __restrict__ kk, const u16* __restrict__ qq,
                       const float* __restrict__ rescale, float* __restrict__ attn,
                       long n, int Cp, int heads, int dh)
{
  int id = blockIdx.x;
  int e = id % dh; id /= dh;
  int d = id % dh; id /= dh;
  int h = id % heads;
  int b = id / heads;
  __shared__ float red[256];
  const int cd = h * dh + d, ce = h * dh + e;
  float s = 0.f;
  for (long r = threadIdx.x; r < n; r += 256) {
    long base = ((long)b * n + r) * Cp;
    s += bf2f(kk[base + cd]) * bf2f(qq[base + ce]);
  }
  red[threadIdx.x] = s; __syncthreads();
  for (int st = 128; st > 0; st >>= 1) {
    if (threadIdx.x < st) red[threadIdx.x] += red[threadIdx.x + st];
    __syncthreads();
  }
  if (threadIdx.x == 0)
    attn[(((long)b * heads + h) * dh + d) * dh + e] = red[0] * rescale[h];
}

__global__ void softmax_rows_k(float* __restrict__ attn, int rows, int dh)
{
  int r = blockIdx.x * blockDim.x + threadIdx.x;
  if (r >= rows) return;
  float* p = attn + (long)r * dh;
  float m = -1e30f;
  for (int i = 0; i < dh; ++i) m = fmaxf(m, p[i]);
  float s = 0.f;
  for (int i = 0; i < dh; ++i) { p[i] = expf(p[i] - m); s += p[i]; }
  float inv = 1.f / s;
  for (int i = 0; i < dh; ++i) p[i] *= inv;
}

// o[b,n,h*dh+d] = sum_e attn[b,h,d,e] * v[b,n,h*dh+e]
__global__ void attn_apply_k(const float* __restrict__ attn, const u16* __restrict__ v,
                             u16* __restrict__ o, long n, int C, int Cp,
                             int heads, int dh)
{
  long idx = (long)blockIdx.x * blockDim.x + threadIdx.x;
  long total = (long)4 * n * C;
  if (idx >= total) return;
  int c = (int)(idx % C);
  long r = idx / C;
  int b = (int)(r / n);
  int h = c / dh, d = c % dh;
  const float* a = attn + (((long)b * heads + h) * dh + d) * dh;
  const u16* vp = v + r * Cp + h * dh;
  float s = 0.f;
  for (int e = 0; e < dh; ++e) s += a[e] * bf2f(vp[e]);
  o[r * Cp + c] = f2bf(s);
}

// layout helpers ------------------------------------------------------
__global__ void pack_nchw_k(const float* __restrict__ x, u16* __restrict__ o,
                            int C, int H, int W, int Cp)
{
  long idx = (long)blockIdx.x * blockDim.x + threadIdx.x;
  long total = (long)4 * H * W * Cp;
  if (idx >= total) return;
  int c = (int)(idx % Cp);
  long r = idx / Cp;
  int xw = (int)(r % W); long r2 = r / W;
  int y = (int)(r2 % H); int b = (int)(r2 / H);
  o[idx] = (c < C) ? f2bf(x[(((long)b * C + c) * H + y) * W + xw]) : (u16)0;
}

__global__ void unpack_add_k(const u16* __restrict__ t, const float* __restrict__ x,
                             float* __restrict__ out, int C, int H, int W, int Cp)
{
  long idx = (long)blockIdx.x * blockDim.x + threadIdx.x;
  long total = (long)4 * C * H * W;
  if (idx >= total) return;
  int xw = (int)(idx % W); long r = idx / W;
  int y = (int)(r % H); r /= H;
  int c = (int)(r % C); int b = (int)(r / C);
  out[idx] = x[idx] + bf2f(t[(((long)b * H + y) * W + xw) * Cp + c]);
}

__global__ void slice_copy_k(const u16* __restrict__ in, u16* __restrict__ out,
                             long rows, int C, int Cpi, int Cpo, int off)
{
  long idx = (long)blockIdx.x * blockDim.x + threadIdx.x;
  long total = rows * C;
  if (idx >= total) return;
  int c = (int)(idx % C);
  long r = idx / C;
  out[r * Cpo + off + c] = in[r * Cpi + c];
}

// =====================================================================
// Host side
// =====================================================================
static inline long cdiv(long a, long b) { return (a + b - 1) / b; }
static inline int r32(int x) { return (x + 31) & ~31; }

struct PW { u16* p; int Kp, Np, taps; };

struct Bump {
  char* base; size_t off;
  void* take(size_t bytes) {
    void* p = base + off;
    off = (off + bytes + 255) & ~(size_t)255;
    return p;
  }
  u16* u(size_t n)  { return (u16*)take(n * sizeof(u16)); }
  float* f(size_t n){ return (float*)take(n * sizeof(float)); }
};

static PW pack_w(hipStream_t s, Bump& ws, const float* w, int K, int N,
                 int taps, int w_si, int w_so)
{
  PW pw; pw.Kp = r32(K); pw.Np = r32(N); pw.taps = taps;
  long total = (long)taps * pw.Np * pw.Kp;
  pw.p = ws.u(total);
  pack_weight_k<<<cdiv(total, 256), 256, 0, s>>>(w, pw.p, K, N, pw.Kp, pw.Np,
                                                 taps, w_si, w_so);
  return pw;
}

static void gemm_conv(hipStream_t s, const u16* in, const PW& pw,
                      const float* bias, const u16* res, u16* out,
                      int Hi, int Wi, int Cin, int Cinp,
                      int Ho, int Wo, int Cout, int Cstride, int coff,
                      int kh, int kw, int stride, int pad, int act, int transposed)
{
  const int Bn = 4;
  const int Mtot = Bn * (transposed ? Hi : Ho) * (transposed ? Wi : Wo);
  dim3 grid((unsigned)cdiv(Mtot, TM), (unsigned)cdiv(Cout, TN));
  const bool fast = (kh == 1 && kw == 1 && stride == 1 && pad == 0 && !transposed);
  if (fast)
    conv_gemm_wmma<0><<<grid, dim3(128), 0, s>>>(in, pw.p, bias, res, out,
        Hi, Wi, Cin, Cinp, Ho, Wo, Cout, Cstride, coff,
        kh, kw, stride, pad, pw.Kp, pw.Np, act);
  else if (transposed)
    conv_gemm_wmma<2><<<grid, dim3(128), 0, s>>>(in, pw.p, bias, res, out,
        Hi, Wi, Cin, Cinp, Ho, Wo, Cout, Cstride, coff,
        kh, kw, stride, pad, pw.Kp, pw.Np, act);
  else
    conv_gemm_wmma<1><<<grid, dim3(128), 0, s>>>(in, pw.p, bias, res, out,
        Hi, Wi, Cin, Cinp, Ho, Wo, Cout, Cstride, coff,
        kh, kw, stride, pad, pw.Kp, pw.Np, act);
}

struct MsabPk {
  PW q, k, v, proj, ff1, ff2;
  const float *rescale, *proj_b, *pos1, *pos2, *ln_g, *ln_b, *ffdw;
};

static MsabPk setup_msab(hipStream_t s, Bump& ws, void* const* d_in, int& i, int C)
{
  const float* wq = (const float*)d_in[i++];
  const float* wk = (const float*)d_in[i++];
  const float* wv = (const float*)d_in[i++];
  MsabPk m;
  m.rescale = (const float*)d_in[i++];
  const float* proj_w = (const float*)d_in[i++];
  m.proj_b = (const float*)d_in[i++];
  m.pos1 = (const float*)d_in[i++];
  m.pos2 = (const float*)d_in[i++];
  m.ln_g = (const float*)d_in[i++];
  m.ln_b = (const float*)d_in[i++];
  const float* ff1 = (const float*)d_in[i++];
  m.ffdw = (const float*)d_in[i++];
  const float* ff2 = (const float*)d_in[i++];
  // wq/wk/wv/proj: [in][out] row-major -> w_si=C, w_so=1
  m.q    = pack_w(s, ws, wq, C, C, 1, C, 1);
  m.k    = pack_w(s, ws, wk, C, C, 1, C, 1);
  m.v    = pack_w(s, ws, wv, C, C, 1, C, 1);
  m.proj = pack_w(s, ws, proj_w, C, C, 1, C, 1);
  // ff1 OIHW (4C,C,1,1): w_si=1, w_so=C ; ff2 OIHW (C,4C,1,1): w_si=1, w_so=4C
  m.ff1 = pack_w(s, ws, ff1, C, 4 * C, 1, 1, C);
  m.ff2 = pack_w(s, ws, ff2, 4 * C, C, 1, 1, 4 * C);
  return m;
}

static void run_msab(hipStream_t s, const MsabPk& P, u16* io,
                     int H, int W, int C, int Cp, int heads,
                     u16* T0, u16* T1, u16* T2, u16* T3,
                     u16* FF1b, u16* FF2b, float* nq, float* nk, float* attn)
{
  const long n = (long)H * W;
  const long rows = 4 * n;
  const int dh = C / heads;   // 31 at every scale

  gemm_conv(s, io, P.q, nullptr, nullptr, T0, H, W, C, Cp, H, W, C, Cp, 0, 1, 1, 1, 0, 0, 0);
  gemm_conv(s, io, P.k, nullptr, nullptr, T1, H, W, C, Cp, H, W, C, Cp, 0, 1, 1, 1, 0, 0, 0);
  gemm_conv(s, io, P.v, nullptr, nullptr, T2, H, W, C, Cp, H, W, C, Cp, 0, 1, 1, 1, 0, 0, 0);

  l2norm_k<<<4 * C, 256, 0, s>>>(T0, nq, n, C, Cp);
  l2norm_k<<<4 * C, 256, 0, s>>>(T1, nk, n, C, Cp);
  long tp = rows * Cp;
  scale_k<<<cdiv(tp, 256), 256, 0, s>>>(T0, nq, n, C, Cp);
  scale_k<<<cdiv(tp, 256), 256, 0, s>>>(T1, nk, n, C, Cp);

  attn_k<<<4 * heads * dh * dh, 256, 0, s>>>(T1, T0, P.rescale, attn, n, Cp, heads, dh);
  int arows = 4 * heads * dh;
  softmax_rows_k<<<cdiv(arows, 64), 64, 0, s>>>(attn, arows, dh);
  attn_apply_k<<<cdiv(rows * C, 256), 256, 0, s>>>(attn, T2, T3, n, C, Cp, heads, dh);

  gemm_conv(s, T3, P.proj, P.proj_b, nullptr, T0, H, W, C, Cp, H, W, C, Cp, 0, 1, 1, 1, 0, 0, 0);

  // positional branch on v + fused adds: io = dw2(gelu(dw1(v))) + proj + io
  dwconv3x3_k<<<cdiv(tp, 256), 256, 0, s>>>(T2, P.pos1, nullptr, nullptr, T1, H, W, C, Cp, 1);
  dwconv3x3_k<<<cdiv(tp, 256), 256, 0, s>>>(T1, P.pos2, T0, io, io, H, W, C, Cp, 0);

  layernorm_k<<<cdiv(rows, 256), 256, 0, s>>>(io, P.ln_g, P.ln_b, T3, rows, C, Cp);
  int C4 = 4 * C, Cp2 = r32(C4);
  gemm_conv(s, T3, P.ff1, nullptr, nullptr, FF1b, H, W, C, Cp, H, W, C4, Cp2, 0, 1, 1, 1, 0, 1, 0);
  dwconv3x3_k<<<cdiv(rows * Cp2, 256), 256, 0, s>>>(FF1b, P.ffdw, nullptr, nullptr, FF2b,
                                                    H, W, C4, Cp2, 1);
  gemm_conv(s, FF2b, P.ff2, nullptr, io, io, H, W, C4, Cp2, H, W, C, Cp, 0, 1, 1, 1, 0, 0, 0);
}

extern "C" void kernel_launch(void* const* d_in, const int* in_sizes, int n_in,
                              void* d_out, int out_size, void* d_ws, size_t ws_size,
                              hipStream_t stream)
{
  (void)in_sizes; (void)n_in; (void)out_size; (void)ws_size;
  hipStream_t s = stream;
  Bump ws{(char*)d_ws, 0};

  const float* x = (const float*)d_in[0];
  int i = 1;
  const float* emb_w = (const float*)d_in[i++];
  MsabPk m_e0 = setup_msab(s, ws, d_in, i, 31);
  const float* down0_w = (const float*)d_in[i++];
  MsabPk m_e1 = setup_msab(s, ws, d_in, i, 62);
  const float* down1_w = (const float*)d_in[i++];
  MsabPk m_bn = setup_msab(s, ws, d_in, i, 124);
  const float* pt_w  = (const float*)d_in[i++];
  const float* pt_b  = (const float*)d_in[i++];
  const float* pt2_w = (const float*)d_in[i++];
  const float* pt2_b = (const float*)d_in[i++];
  const float* u32_w = (const float*)d_in[i++];
  const float* u32_b = (const float*)d_in[i++];
  const float* c2_w  = (const float*)d_in[i++];
  const float* c2_b  = (const float*)d_in[i++];
  MsabPk m_d2 = setup_msab(s, ws, d_in, i, 62);
  const float* u21_w = (const float*)d_in[i++];
  const float* u21_b = (const float*)d_in[i++];
  const float* u31_w = (const float*)d_in[i++];
  const float* u31_b = (const float*)d_in[i++];
  const float* c1_w  = (const float*)d_in[i++];
  MsabPk m_d1 = setup_msab(s, ws, d_in, i, 31);
  const float* map_w = (const float*)d_in[i++];

  // weight packs (OIHW: w_si = taps, w_so = Cin*taps ; convT (Ci,Co,k,k):
  // w_si = Cout*k*k, w_so = k*k)
  PW emb   = pack_w(s, ws, emb_w,   31,  31,  9, 9,   279);
  PW down0 = pack_w(s, ws, down0_w, 31,  62, 16, 16,  496);
  PW down1 = pack_w(s, ws, down1_w, 62, 124, 16, 16,  992);
  PW pt    = pack_w(s, ws, pt_w,    31,  31,  9, 9,   279);
  PW pt2   = pack_w(s, ws, pt2_w,   31,  31,  9, 9,   279);
  PW u32p  = pack_w(s, ws, u32_w,  124,  62,  4, 248, 4);
  PW c2    = pack_w(s, ws, c2_w,   155,  62,  1, 1,   155);
  PW u21p  = pack_w(s, ws, u21_w,   62,  31,  4, 124, 4);
  PW u31p  = pack_w(s, ws, u31_w,  124,  31, 16, 496, 16);
  PW c1    = pack_w(s, ws, c1_w,    93,  31,  1, 1,   93);
  PW mapp  = pack_w(s, ws, map_w,   31,  31,  9, 9,   279);

  const long n1 = 4L * 256 * 256, n2 = 4L * 128 * 128, n3 = 4L * 64 * 64;

  u16* Xn  = ws.u(n1 * 32);
  u16* E0  = ws.u(n1 * 32);
  u16* B2  = ws.u(n2 * 64);
  u16* C3  = ws.u(n3 * 128);
  u16* T0  = ws.u(n1 * 32);
  u16* T1  = ws.u(n1 * 32);
  u16* T2  = ws.u(n1 * 32);
  u16* T3  = ws.u(n1 * 32);
  u16* FF1 = ws.u(n1 * 128);   // hidden: 128 @256^2, 256 @128^2, 512 @64^2
  u16* FF2 = ws.u(n1 * 128);
  u16* G2  = ws.u(n2 * 160);   // concat @128^2: 155 ch pad 160
  u16* G1  = ws.u(n1 * 96);    // concat @256^2: 93 ch pad 96
  u16* F1  = ws.u(n1 * 32);
  float* nq = ws.f(4 * 128);
  float* nk = ws.f(4 * 128);
  float* attn = ws.f(4 * 4 * 31 * 31);

  // 1) pack NCHW fp32 -> NHWC bf16 (pad 32)
  pack_nchw_k<<<cdiv(n1 * 32, 256), 256, 0, s>>>(x, Xn, 31, 256, 256, 32);

  // 2) encoder
  gemm_conv(s, Xn, emb, nullptr, nullptr, E0, 256, 256, 31, 32, 256, 256, 31, 32, 0,
            3, 3, 1, 1, 0, 0);
  run_msab(s, m_e0, E0, 256, 256, 31, 32, 1, T0, T1, T2, T3, FF1, FF2, nq, nk, attn);

  gemm_conv(s, E0, down0, nullptr, nullptr, B2, 256, 256, 31, 32, 128, 128, 62, 64, 0,
            4, 4, 2, 1, 0, 0);
  run_msab(s, m_e1, B2, 128, 128, 62, 64, 2, T0, T1, T2, T3, FF1, FF2, nq, nk, attn);

  gemm_conv(s, B2, down1, nullptr, nullptr, C3, 128, 128, 62, 64, 64, 64, 124, 128, 0,
            4, 4, 2, 1, 0, 0);
  run_msab(s, m_bn, C3, 64, 64, 124, 128, 4, T0, T1, T2, T3, FF1, FF2, nq, nk, attn);

  // 3) skip fusion @128^2 -> G2 = [h1(31) | e1(62) | up32(62)]
  gemm_conv(s, E0, pt, pt_b, nullptr, T0, 256, 256, 31, 32, 128, 128, 31, 32, 0,
            3, 3, 2, 1, 2, 0);                                   // leaky 0.01
  gemm_conv(s, T0, pt2, pt2_b, nullptr, G2, 128, 128, 31, 32, 128, 128, 31, 160, 0,
            3, 3, 1, 1, 0, 0);
  slice_copy_k<<<cdiv(n2 * 62, 256), 256, 0, s>>>(B2, G2, n2, 62, 64, 160, 31);
  gemm_conv(s, C3, u32p, u32_b, nullptr, G2, 64, 64, 124, 128, 128, 128, 62, 160, 93,
            2, 2, 2, 0, 0, 1);                                   // convT 2x2
  gemm_conv(s, G2, c2, c2_b, nullptr, B2, 128, 128, 155, 160, 128, 128, 62, 64, 0,
            1, 1, 1, 0, 0, 0);
  run_msab(s, m_d2, B2, 128, 128, 62, 64, 2, T0, T1, T2, T3, FF1, FF2, nq, nk, attn);

  // 4) skip fusion @256^2 -> G1 = [fea(31) | e0(31) | h3(31)]
  gemm_conv(s, B2, u21p, u21_b, nullptr, G1, 128, 128, 62, 64, 256, 256, 31, 96, 0,
            2, 2, 2, 0, 0, 1);                                   // convT 2x2
  slice_copy_k<<<cdiv(n1 * 31, 256), 256, 0, s>>>(E0, G1, n1, 31, 32, 96, 31);
  gemm_conv(s, C3, u31p, u31_b, nullptr, G1, 64, 64, 124, 128, 256, 256, 31, 96, 62,
            4, 4, 4, 0, 0, 1);                                   // convT 4x4
  gemm_conv(s, G1, c1, nullptr, nullptr, F1, 256, 256, 93, 96, 256, 256, 31, 32, 0,
            1, 1, 1, 0, 0, 0);
  run_msab(s, m_d1, F1, 256, 256, 31, 32, 1, T0, T1, T2, T3, FF1, FF2, nq, nk, attn);

  // 5) mapping conv + residual unpack
  gemm_conv(s, F1, mapp, nullptr, nullptr, T0, 256, 256, 31, 32, 256, 256, 31, 32, 0,
            3, 3, 1, 1, 0, 0);
  unpack_add_k<<<cdiv(4L * 31 * 256 * 256, 256), 256, 0, s>>>(
      T0, x, (float*)d_out, 31, 256, 256, 32);
}
